// Conv2d_14156212208032
// MI455X (gfx1250) — compile-verified, matrix operand order checked by reading, not run on hardware
//
#include <hip/hip_runtime.h>

// ---------------------------------------------------------------------------
// Conv2d 3x3, stride 1, pad 1  (N=32, Cin=128, H=W=56, Cout=256), fp32 in/out.
// Implicit GEMM on CDNA5 WMMA bf16:  D[256, 100352] = W[256,1152] * im2col.
//
// Block = 256 threads (8 wave32). Block tile = 256 (M) x 8x8 spatial pixels.
// A 10x10 halo x 32-channel input slab is staged in LDS once per channel
// chunk; all 9 filter taps read shifted windows from it (9x less global B
// traffic than naive im2col). Weights are pre-reordered to k' = tap*128 + c
// as bf16 so per-lane A fragments are two contiguous b128 loads.
// Each wave: 2 M-subtiles x 4 N-subtiles -> 72 WMMAs per channel chunk.
// ---------------------------------------------------------------------------

typedef __attribute__((ext_vector_type(16))) __bf16 v16bf;
typedef __attribute__((ext_vector_type(8)))  float  v8f;

#define CIN    128
#define COUT   256
#define HH     56
#define WW     56
#define HW     3136        // 56*56
#define NIMG   32
#define KTOT   1152        // 128*9

__device__ __forceinline__ unsigned short f2bf(float f) {
  // round-to-nearest-even f32 -> bf16
  unsigned u = __builtin_bit_cast(unsigned, f);
  u += 0x7FFFu + ((u >> 16) & 1u);
  return (unsigned short)(u >> 16);
}

// Pre-pass: weight[o][c][kh][kw] (f32) -> Aw[o][r*128 + c] (bf16), r = kh*3+kw
__global__ __launch_bounds__(256) void reorder_weights_bf16(
    const float* __restrict__ w, unsigned short* __restrict__ Aw) {
  int idx = blockIdx.x * 256 + threadIdx.x;        // 294912 total
  if (idx >= COUT * KTOT) return;
  int o = idx / KTOT;
  int t = idx - o * KTOT;
  int r = t >> 7;                                  // t / 128
  int c = t & 127;                                 // t % 128
  Aw[idx] = f2bf(w[o * KTOT + c * 9 + r]);
}

union Frag {
  uint4 q[2];
  v16bf v;
};

__global__ __launch_bounds__(256) void conv_wmma_bf16(
    const float* __restrict__ x, const unsigned short* __restrict__ Aw,
    const float* __restrict__ bias, float* __restrict__ out) {
  // 10x10 halo pixels x 32 channels, bf16; rows padded 32->40 ushorts (80 B,
  // 16B-aligned) so half-wave ds_load_b128 fragment reads hit distinct banks.
  __shared__ __align__(16) unsigned short Xs[100][40];   // 8000 B

  const int tid  = threadIdx.x;
  const int wave = tid >> 5;          // 0..7
  const int lane = tid & 31;
  const int half = lane >> 4;         // K-half selector (ISA A/B layout)
  const int col  = lane & 15;         // N col within subtile / M row in A

  // ---- block -> (image, 8x8 spatial tile) : 32 * 7 * 7 = 1568 blocks ----
  const int b   = blockIdx.x;
  const int img = b / 49;
  const int t49 = b - img * 49;
  const int ty  = t49 / 7;
  const int tx  = t49 - ty * 7;
  const int by  = ty * 8;
  const int bx  = tx * 8;
  const float* xImg = x + (size_t)img * (CIN * HW);

  // ---- halo fill mapping: element e = (haloPixel, 8-channel group) ----
  // 100 halo pixels * 4 groups = 400 elements; thread does e0=tid, e1=tid+256.
  const int hp0 = tid >> 2,        cg0 = tid & 3;
  const int hp1 = (tid + 256) >> 2, cg1 = tid & 3;   // (tid+256)&3 == tid&3
  const int hy0 = hp0 / 10, hx0 = hp0 - hy0 * 10;
  const int hy1 = hp1 / 10, hx1 = hp1 - hy1 * 10;
  const int ih0 = by + hy0 - 1, iw0 = bx + hx0 - 1;
  const int ih1 = by + hy1 - 1, iw1 = bx + hx1 - 1;
  const bool w1 = (tid < 144);                       // second element exists
  const bool v0 = ((unsigned)ih0 < HH) && ((unsigned)iw0 < WW);
  const bool v1 = w1 && ((unsigned)ih1 < HH) && ((unsigned)iw1 < WW);
  const float* s0 = xImg + (v0 ? (ih0 * WW + iw0) : 0);
  const float* s1 = xImg + (v1 ? (ih1 * WW + iw1) : 0);

  // ---- per-wave M subtiles ----
  const int mT0 = wave * 16;          // 0..112
  const int mT1 = mT0 + 128;          // 128..240

  v8f acc[8] = {};                    // [mi*4 + ns]
  float pf[16];                       // prefetched halo values

  auto loadHalo = [&](int cc) {
    const int cB = cc << 5;
#pragma unroll
    for (int j = 0; j < 8; ++j) {
      pf[j]     = v0 ? s0[(size_t)(cB + cg0 * 8 + j) * HW] : 0.0f;
      pf[8 + j] = v1 ? s1[(size_t)(cB + cg1 * 8 + j) * HW] : 0.0f;
    }
  };

  loadHalo(0);
  for (int cc = 0; cc < 4; ++cc) {    // four 32-channel chunks
    __syncthreads();                  // previous chunk's Xs reads done
    {
      union { unsigned short s[8]; uint4 v; } pk;
#pragma unroll
      for (int j = 0; j < 8; ++j) pk.s[j] = f2bf(pf[j]);
      *(uint4*)(&Xs[hp0][cg0 * 8]) = pk.v;
      if (w1) {
#pragma unroll
        for (int j = 0; j < 8; ++j) pk.s[j] = f2bf(pf[8 + j]);
        *(uint4*)(&Xs[hp1][cg1 * 8]) = pk.v;
      }
    }
    __syncthreads();                  // halo slab ready
    if (cc < 3) loadHalo(cc + 1);     // overlap next gather with 72 WMMAs

    const int cB = cc << 5;
#pragma unroll
    for (int r = 0; r < 9; ++r) {     // filter taps from shifted LDS windows
      const int kh = r / 3;
      const int kw = r - kh * 3;
      const int kg = r * 128 + cB + half * 8;   // reordered global K

      Frag a0, a1;
      {
        const unsigned short* p0 = Aw + (size_t)(mT0 + col) * KTOT + kg;
        const unsigned short* p1 = Aw + (size_t)(mT1 + col) * KTOT + kg;
        a0.q[0] = *(const uint4*)(p0);
        a0.q[1] = *(const uint4*)(p0 + 16);
        a1.q[0] = *(const uint4*)(p1);
        a1.q[1] = *(const uint4*)(p1 + 16);
      }

#pragma unroll
      for (int ns = 0; ns < 4; ++ns) {
        const int n  = ns * 16 + col;                       // pixel in tile
        const int rp = ((n >> 3) + kh) * 10 + (n & 7) + kw; // halo pixel
        Frag bq;
        bq.q[0] = *(const uint4*)(&Xs[rp][half * 8]);       // k = half*8..
        bq.q[1] = *(const uint4*)(&Xs[rp][16 + half * 8]);  // k = 16+half*8..
        acc[ns] = __builtin_amdgcn_wmma_f32_16x16x32_bf16(
            false, a0.v, false, bq.v, (short)0, acc[ns], false, false);
        acc[4 + ns] = __builtin_amdgcn_wmma_f32_16x16x32_bf16(
            false, a1.v, false, bq.v, (short)0, acc[4 + ns], false, false);
      }
    }
  }

  // ---- epilogue: D layout VGPR j -> M = j + 8*half, lane&15 -> N ----
#pragma unroll
  for (int ns = 0; ns < 4; ++ns) {
    const int n  = ns * 16 + col;
    const int oh = by + (n >> 3);
    const int ow = bx + (n & 7);
    float* ob = out + (size_t)img * (COUT * HW) + oh * WW + ow;
#pragma unroll
    for (int j = 0; j < 8; ++j) {
      const int mA = mT0 + j + 8 * half;
      const int mB = mT1 + j + 8 * half;
      ob[(size_t)mA * HW] = acc[ns][j] + bias[mA];
      ob[(size_t)mB * HW] = acc[4 + ns][j] + bias[mB];
    }
  }
}

extern "C" void kernel_launch(void* const* d_in, const int* in_sizes, int n_in,
                              void* d_out, int out_size, void* d_ws, size_t ws_size,
                              hipStream_t stream) {
  const float* x    = (const float*)d_in[0];  // (32,128,56,56)
  const float* wgt  = (const float*)d_in[1];  // (256,128,3,3)
  const float* bias = (const float*)d_in[2];  // (256,)
  float* out        = (float*)d_out;          // (32,256,56,56)
  unsigned short* Aw = (unsigned short*)d_ws; // 256*1152 bf16 = 576 KB

  (void)in_sizes; (void)n_in; (void)out_size; (void)ws_size;

  reorder_weights_bf16<<<(COUT * KTOT + 255) / 256, 256, 0, stream>>>(wgt, Aw);
  conv_wmma_bf16<<<NIMG * 49, 256, 0, stream>>>(x, Aw, bias, out);
}
